// SFormer_53223234732618
// MI455X (gfx1250) — matrix operand order repaired and checked by reading, not compiled, not run on hardware
//
#include <hip/hip_runtime.h>

// Problem constants (from reference): T=64, k1=256, k2=1024, D=1024
constexpr int TT  = 64;
constexpr int KK1 = 256;
constexpr int KK2 = 1024;
constexpr int DD  = 1024;

typedef __attribute__((ext_vector_type(16))) __bf16 v16bf;
typedef __attribute__((ext_vector_type(8)))  __bf16 v8bf;
typedef __attribute__((ext_vector_type(8)))  float  v8f;

__device__ __forceinline__ v16bf cat8(v8bf lo, v8bf hi) {
  return __builtin_shufflevector(lo, hi, 0,1,2,3,4,5,6,7,8,9,10,11,12,13,14,15);
}

__device__ __forceinline__ v8f wmma_bf16(v16bf a, v16bf b, v8f c) {
  // emits v_wmma_f32_16x16x32_bf16
  return __builtin_amdgcn_wmma_f32_16x16x32_bf16(false, a, false, b, (short)0, c, false, false);
}

// ---------------------------------------------------------------------------
// Convert slow_feature -> bf16 row-major [t][k][d] and transposed [t][d][k]
// ---------------------------------------------------------------------------
__global__ __launch_bounds__(256) void cvt_slow(const float* __restrict__ slow,
                                                __bf16* __restrict__ sf16,
                                                __bf16* __restrict__ sfT16) {
  size_t idx = (size_t)blockIdx.x * 256 + threadIdx.x;
  int t   = (int)(idx / (KK1 * DD));
  int rem = (int)(idx % (KK1 * DD));
  int k = rem / DD;
  int d = rem % DD;
  __bf16 v = (__bf16)slow[idx];
  sf16[idx] = v;
  sfT16[((size_t)t * DD + d) * KK1 + k] = v;
}

// Convert fast_W (transposed: WfT[d][e] = fW[e][d]) and slow_W (as-is) to bf16
__global__ __launch_bounds__(256) void cvt_w(const float* __restrict__ fW,
                                             const float* __restrict__ sW,
                                             __bf16* __restrict__ WfT16,
                                             __bf16* __restrict__ Ws16) {
  int idx = blockIdx.x * 256 + threadIdx.x;   // idx = e*DD + d
  int e = idx / DD, d = idx % DD;
  WfT16[d * DD + e] = (__bf16)fW[idx];
  Ws16[idx]         = (__bf16)sW[idx];
}

// ---------------------------------------------------------------------------
// Generic bf16 WMMA GEMM:  Out[M][N] = A[M][K] * B[N][K]^T (+ bias[N])
// Block = 4 waves, each wave does a 16x64 tile; block tile = 16 x 256.
// ---------------------------------------------------------------------------
__global__ __launch_bounds__(128) void gemm_bf16(const __bf16* __restrict__ A,
                                                 const __bf16* __restrict__ B,
                                                 const float*  __restrict__ bias,
                                                 __bf16* __restrict__ Out,
                                                 int M, int N, int K) {
  const int lane = threadIdx.x & 31;
  const int wave = threadIdx.x >> 5;
  const int row0 = blockIdx.x * 16;
  const int col0 = blockIdx.y * 256 + wave * 64;
  const int arow = lane & 15;
  const int hi   = (lane >> 4) & 1;

  const __bf16* aRow = A + (size_t)(row0 + arow) * K;

  v8f zero = {0.f,0.f,0.f,0.f,0.f,0.f,0.f,0.f};
  v8f acc[4];
#pragma unroll
  for (int i = 0; i < 4; ++i) acc[i] = zero;

  for (int kk = 0; kk < K; kk += 32) {
    // A-operand layout: lane-lo holds K {0..7,16..23}, lane-hi {8..15,24..31}
    v8bf a0 = *(const v8bf*)(aRow + kk + hi * 8);
    v8bf a1 = *(const v8bf*)(aRow + kk + hi * 8 + 16);
    v16bf a = cat8(a0, a1);
#pragma unroll
    for (int nt = 0; nt < 4; ++nt) {
      // B-operand layout: lane-lo holds K {0..7,16..23}, lane-hi {8..15,24..31}
      const __bf16* bRow = B + (size_t)(col0 + nt * 16 + arow) * K;
      v16bf b = cat8(*(const v8bf*)(bRow + kk + hi * 8), *(const v8bf*)(bRow + kk + hi * 8 + 16));
      acc[nt] = wmma_bf16(a, b, acc[nt]);
    }
  }

  const int mbase = row0 + hi * 8;
#pragma unroll
  for (int nt = 0; nt < 4; ++nt) {
    int n = col0 + nt * 16 + arow;
    float bv = bias ? bias[n] : 0.f;
#pragma unroll
    for (int v = 0; v < 8; ++v)
      Out[(size_t)(mbase + v) * N + n] = (__bf16)(acc[nt][v] + bv);
  }
}

// ---------------------------------------------------------------------------
// bterm[r] = fast_b . slow_key16[r]   (one wave per row)
// ---------------------------------------------------------------------------
__global__ __launch_bounds__(256) void bterm_k(const __bf16* __restrict__ sk,
                                               const float* __restrict__ fb,
                                               float* __restrict__ bterm) {
  int row  = blockIdx.x * 8 + (threadIdx.x >> 5);
  int lane = threadIdx.x & 31;
  const __bf16* p = sk + (size_t)row * DD;
  float s = 0.f;
  for (int e = lane; e < DD; e += 32) s += (float)p[e] * fb[e];
#pragma unroll
  for (int m = 16; m; m >>= 1) s += __shfl_xor(s, m, 32);
  if (lane == 0) bterm[row] = s;
}

// ---------------------------------------------------------------------------
// Fused attention: sim = fast @ Z^T + bterm, softmax, ctx = attn @ slow, +res
// Grid: (T, KK2/64). Block: 128 threads = 4 waves; wave w -> rows q0..q0+15.
// ---------------------------------------------------------------------------
__global__ __launch_bounds__(128) void attn_fused(const float*  __restrict__ fast,
                                                  const __bf16* __restrict__ Z,     // [T][KK1][DD]
                                                  const __bf16* __restrict__ sfT,   // [T][DD][KK1]
                                                  const float*  __restrict__ bterm, // [T][KK1]
                                                  float* __restrict__ out) {
  __shared__ __bf16 zbuf[KK1][32];        // 16 KB : staged Z k-slab, shared by all waves
  __shared__ __bf16 albuf[4][16][KK1];    // 32 KB : per-wave attention tile (C->A relayout)

  const int t    = blockIdx.x;
  const int lane = threadIdx.x & 31;
  const int wave = threadIdx.x >> 5;
  const int q0   = blockIdx.y * 64 + wave * 16;
  const int arow = lane & 15;
  const int hi   = (lane >> 4) & 1;
  const int b8   = hi * 8;
  const int mb   = hi * 8;

  v8f zero = {0.f,0.f,0.f,0.f,0.f,0.f,0.f,0.f};
  v8f acc[16];
#pragma unroll
  for (int i = 0; i < 16; ++i) acc[i] = zero;

  const float*  fRow = fast + ((size_t)t * KK2 + q0 + arow) * DD;
  const __bf16* Zt   = Z + (size_t)t * KK1 * DD;

  // ---- Phase 1: sim = fast @ Z^T (f32 accum), K loop over D ----
  for (int kk = 0; kk < DD; kk += 32) {
    __syncthreads();
    for (int r = threadIdx.x; r < KK1; r += 128) {
      const uint4* s = (const uint4*)(Zt + (size_t)r * DD + kk);
      uint4* dst = (uint4*)(&zbuf[r][0]);
      dst[0] = s[0]; dst[1] = s[1]; dst[2] = s[2]; dst[3] = s[3];
    }
    __syncthreads();

    // A tile: convert fp32 fast_feature -> bf16 in-register
    const float* ap = fRow + kk + b8;
    float4 f0 = *(const float4*)(ap);
    float4 f1 = *(const float4*)(ap + 4);
    float4 f2 = *(const float4*)(ap + 16);
    float4 f3 = *(const float4*)(ap + 20);
    v16bf a;
    a[0]=(__bf16)f0.x;  a[1]=(__bf16)f0.y;  a[2]=(__bf16)f0.z;  a[3]=(__bf16)f0.w;
    a[4]=(__bf16)f1.x;  a[5]=(__bf16)f1.y;  a[6]=(__bf16)f1.z;  a[7]=(__bf16)f1.w;
    a[8]=(__bf16)f2.x;  a[9]=(__bf16)f2.y;  a[10]=(__bf16)f2.z; a[11]=(__bf16)f2.w;
    a[12]=(__bf16)f3.x; a[13]=(__bf16)f3.y; a[14]=(__bf16)f3.z; a[15]=(__bf16)f3.w;

#pragma unroll
    for (int nt = 0; nt < 16; ++nt) {
      v16bf b = cat8(*(const v8bf*)(&zbuf[nt * 16 + arow][hi * 8]), *(const v8bf*)(&zbuf[nt * 16 + arow][hi * 8 + 16]));
      acc[nt] = wmma_bf16(a, b, acc[nt]);
    }
  }

  // ---- bias + 1/sqrt(D) scale ----
  {
    const float* bt = bterm + t * KK1;
#pragma unroll
    for (int nt = 0; nt < 16; ++nt) {
      float bv = bt[nt * 16 + arow];
#pragma unroll
      for (int v = 0; v < 8; ++v) acc[nt][v] = (acc[nt][v] + bv) * 0.03125f;
    }
  }

  // ---- softmax over k: each row lives in one 16-lane half x 16 tiles ----
#pragma unroll
  for (int v = 0; v < 8; ++v) {
    float mx = acc[0][v];
#pragma unroll
    for (int nt = 1; nt < 16; ++nt) mx = fmaxf(mx, acc[nt][v]);
    mx = fmaxf(mx, __shfl_xor(mx, 1, 16));
    mx = fmaxf(mx, __shfl_xor(mx, 2, 16));
    mx = fmaxf(mx, __shfl_xor(mx, 4, 16));
    mx = fmaxf(mx, __shfl_xor(mx, 8, 16));
    float sum = 0.f;
#pragma unroll
    for (int nt = 0; nt < 16; ++nt) {
      float e = __expf(acc[nt][v] - mx);
      acc[nt][v] = e;
      sum += e;
    }
    sum += __shfl_xor(sum, 1, 16);
    sum += __shfl_xor(sum, 2, 16);
    sum += __shfl_xor(sum, 4, 16);
    sum += __shfl_xor(sum, 8, 16);
    float inv = 1.f / sum;
#pragma unroll
    for (int nt = 0; nt < 16; ++nt) acc[nt][v] *= inv;
  }

  // ---- re-layout attn C->A through per-wave LDS (bf16) ----
#pragma unroll
  for (int nt = 0; nt < 16; ++nt)
#pragma unroll
    for (int v = 0; v < 8; ++v)
      albuf[wave][mb + v][nt * 16 + arow] = (__bf16)acc[nt][v];

  // ---- Phase 2: ctx = attn @ slow_feature, residual add in fp32 ----
  const __bf16* sfTt = sfT + (size_t)t * DD * KK1;
  for (int d0 = 0; d0 < DD; d0 += 64) {
    v8f c[4];
#pragma unroll
    for (int i = 0; i < 4; ++i) c[i] = zero;
    for (int k8 = 0; k8 < KK1; k8 += 32) {
      v8bf a0 = *(const v8bf*)(&albuf[wave][arow][k8 + b8]);
      v8bf a1 = *(const v8bf*)(&albuf[wave][arow][k8 + b8 + 16]);
      v16bf a = cat8(a0, a1);
#pragma unroll
      for (int nt = 0; nt < 4; ++nt) {
        int n = d0 + nt * 16 + arow;
        v16bf b = cat8(*(const v8bf*)(sfTt + (size_t)n * KK1 + k8 + hi * 8), *(const v8bf*)(sfTt + (size_t)n * KK1 + k8 + hi * 8 + 16));
        c[nt] = wmma_bf16(a, b, c[nt]);
      }
    }
#pragma unroll
    for (int nt = 0; nt < 4; ++nt) {
      int n = d0 + nt * 16 + arow;
#pragma unroll
      for (int v = 0; v < 8; ++v) {
        size_t idx = ((size_t)t * KK2 + q0 + mb + v) * DD + n;
        out[idx] = fast[idx] + c[nt][v];
      }
    }
  }
}

// ---------------------------------------------------------------------------
// Launch
// ---------------------------------------------------------------------------
extern "C" void kernel_launch(void* const* d_in, const int* in_sizes, int n_in,
                              void* d_out, int out_size, void* d_ws, size_t ws_size,
                              hipStream_t stream) {
  (void)in_sizes; (void)n_in; (void)out_size; (void)ws_size;
  const float* fast = (const float*)d_in[0];
  const float* slow = (const float*)d_in[1];
  const float* fW   = (const float*)d_in[2];
  const float* fb   = (const float*)d_in[3];
  const float* sW   = (const float*)d_in[4];
  const float* sb   = (const float*)d_in[5];
  float* out = (float*)d_out;

  constexpr size_t SFB = (size_t)TT * KK1 * DD * 2;  // 32 MB
  constexpr size_t WB  = (size_t)DD * DD * 2;        // 2 MB
  char* ws = (char*)d_ws;
  __bf16* sf16   = (__bf16*)(ws);                 // slow_feature bf16 [t][k][d]
  __bf16* sfT16  = (__bf16*)(ws + SFB);           // slow_feature bf16 [t][d][k]
  __bf16* sk16   = (__bf16*)(ws + 2 * SFB);       // slow_key bf16     [t][k][e]
  __bf16* WfT16  = (__bf16*)(ws + 3 * SFB);       // fast_W^T bf16     [d][e]
  __bf16* Ws16   = (__bf16*)(ws + 3 * SFB + WB);  // slow_W bf16       [e][d]
  float*  btermp = (float*)(ws + 3 * SFB + 2 * WB);
  __bf16* Z16 = sf16;  // Z aliases sf16 (sf16 dead after slow_key GEMM)

  cvt_slow<<<(TT * KK1 * DD) / 256, 256, 0, stream>>>(slow, sf16, sfT16);
  cvt_w<<<(DD * DD) / 256, 256, 0, stream>>>(fW, sW, WfT16, Ws16);

  dim3 gg((TT * KK1) / 16, DD / 256);
  // slow_key = slow_feature @ slow_W^T + slow_b
  gemm_bf16<<<gg, 128, 0, stream>>>(sf16, Ws16, sb, sk16, TT * KK1, DD, DD);
  // bterm = slow_key . fast_b
  bterm_k<<<(TT * KK1) / 8, 256, 0, stream>>>(sk16, fb, btermp);
  // Z = slow_key @ fast_W  (B = fast_W^T in N-major layout)
  gemm_bf16<<<gg, 128, 0, stream>>>(sk16, WfT16, nullptr, Z16, TT * KK1, DD, DD);
  // fused attention + residual
  attn_fused<<<dim3(TT, KK2 / 64), 128, 0, stream>>>(fast, Z16, sfT16, btermp, out);
}
